// KNN_torch_5815385719465
// MI455X (gfx1250) — matrix operand order repaired and checked by reading, compiled, not run on hardware
//
#include <hip/hip_runtime.h>
#include <hip/hip_bf16.h>

typedef _Float16 v16h __attribute__((ext_vector_type(16)));
typedef _Float16 v8h  __attribute__((ext_vector_type(8)));
typedef float    v8f  __attribute__((ext_vector_type(8)));
typedef unsigned int u32x4 __attribute__((ext_vector_type(4)));
typedef int          i32x8 __attribute__((ext_vector_type(8)));
typedef int          i32x4 __attribute__((ext_vector_type(4)));

#define DIM         256
#define KNN_SPLIT   32
#define TPITCH      264                     // 256 + 8 halves pad -> conflict-free LDS
#define CHUNK_ROWS  32                      // candidates per iteration (2 WMMA n-tiles)
#define CHUNK_BYTES (CHUNK_ROWS * TPITCH * 2)   // 16896 B per LDS chunk buffer
#define KINV        14.426950408889634f     // 1/(T*ln2), T = 0.1
#define DSENT       1.0e30f

#if __has_builtin(__builtin_amdgcn_tensor_load_to_lds)
#define USE_TDM 1
#else
#define USE_TDM 0
#endif

#if __has_builtin(__builtin_amdgcn_sched_barrier)
#define SCHED_BARRIER() __builtin_amdgcn_sched_barrier(0)
#else
#define SCHED_BARRIER()
#endif

__device__ __forceinline__ void wait_tensor0() {
#if __has_builtin(__builtin_amdgcn_s_wait_tensorcnt)
  __builtin_amdgcn_s_wait_tensorcnt(0);
#else
  asm volatile("s_wait_tensorcnt 0x0" ::: "memory");
#endif
}

#if USE_TDM
// One TDM descriptor moves a 32x512B tile (row pitch 512B in memory) into LDS,
// inserting 16B of padding after every 512B row -> LDS pitch 528B == TPITCH halves.
__device__ __forceinline__ void tdm_load_chunk(const _Float16* tB, int chunk,
                                               unsigned lds_byte) {
  unsigned long long ga =
      (unsigned long long)(const void*)(tB + (size_t)chunk * CHUNK_ROWS * DIM);
  u32x4 g0;
  g0[0] = 1u;                                    // count=1, user desc, no gather
  g0[1] = lds_byte;                              // lds_addr
  g0[2] = (unsigned)(ga & 0xFFFFFFFFu);          // global_addr[31:0]
  g0[3] = (unsigned)((ga >> 32) & 0x1FFFFFFu)    // global_addr[56:32]
        | (2u << 30);                            // type = 2 ("image")
  i32x8 g1;
  g1[0] = (3 << 16)              // data_size = 8B
        | (1 << 20)              // pad_enable
        | (6 << 22)              // pad_interval: 128 DWORDs (one 512B row)
        | (3 << 25);             // pad_amount: 4 DWORDs (16B)
  g1[1] = (64 << 16);            // tensor_dim0 = 64 (8B units, bits[79:48])
  g1[2] = (CHUNK_ROWS << 16);    // tensor_dim1 = 32 (bits[111:80])
  g1[3] = (64 << 16);            // tile_dim0 = 64 (bits[127:112])
  g1[4] = CHUNK_ROWS;            // tile_dim1 = 32 (bits[143:128]); tile_dim2 = 0
  g1[5] = 64;                    // tensor_dim0_stride = 64 (bits[207:160])
  g1[6] = 0;
  g1[7] = 0;                     // tensor_dim1_stride = 0 (2D)
  i32x4 gz4 = {0, 0, 0, 0};
  i32x8 gz8 = {0, 0, 0, 0, 0, 0, 0, 0};
  __builtin_amdgcn_tensor_load_to_lds(g0, g1, gz4, gz4, gz8, 0);
}
#endif

// ---------------- kernel 1: L2-normalize rows, f32 -> f16 ----------------
__global__ __launch_bounds__(256) void knn_norm_f16(
    const float* __restrict__ src, _Float16* __restrict__ dst, int rows)
{
  const int wave = threadIdx.x >> 5, lane = threadIdx.x & 31;
  const int row = blockIdx.x * 8 + wave;
  if (row >= rows) return;
  const float4* p = (const float4*)(src + (size_t)row * DIM) + lane * 2;
  float4 v0 = p[0], v1 = p[1];
  float ss = v0.x*v0.x + v0.y*v0.y + v0.z*v0.z + v0.w*v0.w
           + v1.x*v1.x + v1.y*v1.y + v1.z*v1.z + v1.w*v1.w;
#pragma unroll
  for (int m = 1; m < 32; m <<= 1) ss += __shfl_xor(ss, m, 32);
  const float inv = 1.0f / fmaxf(sqrtf(ss), 1e-12f);
  v8h o;
  o[0]=(_Float16)(v0.x*inv); o[1]=(_Float16)(v0.y*inv);
  o[2]=(_Float16)(v0.z*inv); o[3]=(_Float16)(v0.w*inv);
  o[4]=(_Float16)(v1.x*inv); o[5]=(_Float16)(v1.y*inv);
  o[6]=(_Float16)(v1.z*inv); o[7]=(_Float16)(v1.w*inv);
  *(v8h*)(dst + (size_t)row * DIM + lane * 8) = o;
}

// ---------------- kernel 2: fused WMMA GEMM + online softmax ----------------
__global__ __launch_bounds__(256) void knn_fused(
    const _Float16* __restrict__ qA, const _Float16* __restrict__ tB,
    float4* __restrict__ partial, int B, int nChunks, int per)
{
  __shared__ _Float16 smem[2 * CHUNK_ROWS * TPITCH];
  const int tid  = threadIdx.x;
  const int wave = tid >> 5, lane = tid & 31;
  const int m = lane & 15, h = lane >> 4;
  const int g = blockIdx.y;
  const int start = g * per;
  const int end   = min(start + per, nChunks);
  const int qtile = blockIdx.x * 8 + wave;
  const bool active = (qtile * 16) < B;
  const int qbase = (active ? qtile : 0) * 16;

  if (start >= end) {                      // block-uniform: no barrier hazard
    if (active && h == 0)
      partial[(size_t)(qbase + m) * KNN_SPLIT + g] = make_float4(DSENT, 0.f, 0.f, 0.f);
    return;
  }

  // A fragments: lane(m,h): a[i]=q[m][32c+16h+i]
  v16h a[8];
  const _Float16* qrow = qA + (size_t)(qbase + m) * DIM;
#pragma unroll
  for (int c = 0; c < 8; ++c) {
    v8h lo = *(const v8h*)(qrow + c * 32 + 16 * h);
    v8h hi = *(const v8h*)(qrow + c * 32 + 16 * h + 8);
    a[c] = __builtin_shufflevector(lo, hi, 0,1,2,3,4,5,6,7,8,9,10,11,12,13,14,15);
  }

  float d0[8], se[8], wd[8];
#pragma unroll
  for (int r = 0; r < 8; ++r) { d0[r] = DSENT; se[r] = 0.f; wd[r] = 0.f; }

  const unsigned lds_base = (unsigned)(size_t)(void*)smem;

#if !USE_TDM
  auto load_chunk = [&](int chunk, int buf) {
    // 256 threads copy 32x512B; two rows per thread-slab
#pragma unroll
    for (int rr = 0; rr < 2; ++rr) {
      const int row = (tid >> 4) + rr * 16, seg = tid & 15;
      const _Float16* s = tB + ((size_t)chunk * CHUNK_ROWS + row) * DIM + seg * 16;
      _Float16* d = &smem[buf * CHUNK_ROWS * TPITCH + row * TPITCH + seg * 16];
      v8h x0 = *(const v8h*)s;
      v8h x1 = *(const v8h*)(s + 8);
      *(v8h*)d = x0;
      *(v8h*)(d + 8) = x1;
    }
  };
#endif

#if USE_TDM
  if (wave == 0) { tdm_load_chunk(tB, start, lds_base); wait_tensor0(); }
#else
  load_chunk(start, 0);
#endif
  __syncthreads();

  // online-softmax slot update (r is a compile-time constant at each call site)
  auto upd = [&](int r, float dot) {
    float dist = __builtin_amdgcn_sqrtf(fmaxf(2.0f - 2.0f * dot, 0.0f));
    if (dist < d0[r]) {
      float al = __builtin_amdgcn_exp2f((dist - d0[r]) * KINV);
      se[r] = se[r] * al + 1.0f;
      wd[r] = wd[r] * al + dist;
      d0[r] = dist;
    } else {
      float pw = __builtin_amdgcn_exp2f((d0[r] - dist) * KINV);
      se[r] += pw;
      wd[r] += dist * pw;
    }
  };

  for (int i = start; i < end; ++i) {
    const int cb = (i - start) & 1;

#if USE_TDM
    if (wave == 0) {                       // DMA next 32-row chunk into back buffer
      if (i + 1 < end)
        tdm_load_chunk(tB, i + 1, lds_base + (unsigned)((cb ^ 1) * CHUNK_BYTES));
    } else if (wave == 1 && i + 2 < end) { // warm L2 two chunks ahead
      __builtin_prefetch(tB + ((size_t)(i + 2) * CHUNK_ROWS) * DIM + lane * 512, 0, 1);
    }
#else
    if (i + 1 < end) load_chunk(i + 1, cb ^ 1);
#endif

    const _Float16* brow = &smem[cb * CHUNK_ROWS * TPITCH + m * TPITCH];

    // ---- n-tile 0 (chunk rows 0..15): batch all 16 ds_load_b128 first ----
    v16h b[8];
#pragma unroll
    for (int c = 0; c < 8; ++c) {
      v8h b0 = *(const v8h*)(brow + c * 32 + 16 * h);
      v8h b1 = *(const v8h*)(brow + c * 32 + 16 * h + 8);
      b[c] = __builtin_shufflevector(b0, b1, 0,1,2,3,4,5,6,7,8,9,10,11,12,13,14,15);
    }
    SCHED_BARRIER();                       // keep loads above the WMMA chain

    v8f acc0 = {0.f,0.f,0.f,0.f,0.f,0.f,0.f,0.f};
#pragma unroll
    for (int c = 0; c < 8; ++c)
      acc0 = __builtin_amdgcn_wmma_f32_16x16x32_f16(
          false, a[c], false, b[c], (short)0, acc0, false, false);

    // ---- n-tile 1 (chunk rows 16..31): loads interleave with chain 0 ----
    const _Float16* brow1 = brow + 16 * TPITCH;
    v16h b2[8];
#pragma unroll
    for (int c = 0; c < 8; ++c) {
      v8h b0 = *(const v8h*)(brow1 + c * 32 + 16 * h);
      v8h b1 = *(const v8h*)(brow1 + c * 32 + 16 * h + 8);
      b2[c] = __builtin_shufflevector(b0, b1, 0,1,2,3,4,5,6,7,8,9,10,11,12,13,14,15);
    }
    SCHED_BARRIER();

    v8f acc1 = {0.f,0.f,0.f,0.f,0.f,0.f,0.f,0.f};
#pragma unroll
    for (int c = 0; c < 8; ++c)
      acc1 = __builtin_amdgcn_wmma_f32_16x16x32_f16(
          false, a[c], false, b2[c], (short)0, acc1, false, false);

    // slot r = query m=r+8h; acc0: n=i*32+(lane&15), acc1: n=i*32+16+(lane&15)
#pragma unroll
    for (int r = 0; r < 8; ++r) { upd(r, acc0[r]); upd(r, acc1[r]); }

#if USE_TDM
    if (wave == 0) wait_tensor0();         // next chunk landed before we signal
#endif
    __syncthreads();
  }

  // butterfly merge across the 16 lanes sharing this query set (same h)
#pragma unroll
  for (int msk = 1; msk < 16; msk <<= 1) {
#pragma unroll
    for (int r = 0; r < 8; ++r) {
      float od = __shfl_xor(d0[r], msk, 32);
      float os = __shfl_xor(se[r], msk, 32);
      float ow = __shfl_xor(wd[r], msk, 32);
      float dn = fminf(d0[r], od);
      float fa = __builtin_amdgcn_exp2f((dn - d0[r]) * KINV);
      float fb = __builtin_amdgcn_exp2f((dn - od)   * KINV);
      se[r] = se[r] * fa + os * fb;
      wd[r] = wd[r] * fa + ow * fb;
      d0[r] = dn;
    }
  }

  if (active && m == 0) {                  // lanes 0 and 16 write 8 queries each
#pragma unroll
    for (int r = 0; r < 8; ++r) {
      const int q = qbase + r + 8 * h;
      partial[(size_t)q * KNN_SPLIT + g] = make_float4(d0[r], se[r], wd[r], 0.f);
    }
  }
}

// ---------------- kernel 3: merge split-N partials ----------------
__global__ __launch_bounds__(256) void knn_combine(
    const float4* __restrict__ partial, float* __restrict__ out, int B)
{
  const int q = blockIdx.x * 256 + threadIdx.x;
  if (q >= B) return;
  float d0 = DSENT, se = 0.f, wd = 0.f;
#pragma unroll 4
  for (int g = 0; g < KNN_SPLIT; ++g) {
    float4 p = partial[(size_t)q * KNN_SPLIT + g];
    float dn = fminf(d0, p.x);
    float fa = exp2f((dn - d0)  * KINV);
    float fb = exp2f((dn - p.x) * KINV);
    se = se * fa + p.y * fb;
    wd = wd * fa + p.z * fb;
    d0 = dn;
  }
  out[q] = wd / se;
}

// ---------------- launcher ----------------
// ws layout: [ qA f16 : B*256*2 ][ tB f16 : N*256*2 ][ partial float4 : B*KNN_SPLIT*16 ]
// total for B=1024, N=100000: ~52.3 MB
extern "C" void kernel_launch(void* const* d_in, const int* in_sizes, int n_in,
                              void* d_out, int out_size, void* d_ws, size_t ws_size,
                              hipStream_t stream)
{
  const float* q32 = (const float*)d_in[0];   // latent_eval  [B,256] f32
  const float* t32 = (const float*)d_in[1];   // train_latents[N,256] f32
  float* out = (float*)d_out;                 // [B] f32
  const int B = in_sizes[0] / DIM;
  const int N = in_sizes[1] / DIM;

  char* ws = (char*)d_ws;
  _Float16* qA = (_Float16*)ws;
  _Float16* tB = (_Float16*)(ws + (size_t)B * DIM * sizeof(_Float16));
  float4* partial = (float4*)(ws + (size_t)(B + N) * DIM * sizeof(_Float16));

  knn_norm_f16<<<(B + 7) / 8, 256, 0, stream>>>(q32, qA, B);
  knn_norm_f16<<<(N + 7) / 8, 256, 0, stream>>>(t32, tB, N);

  const int nChunks = N / CHUNK_ROWS;                   // N = 100000 -> 3125 chunks
  const int per = (nChunks + KNN_SPLIT - 1) / KNN_SPLIT;
  dim3 grid((B + 127) / 128, KNN_SPLIT);
  knn_fused<<<grid, 256, 0, stream>>>(qA, tB, partial, B, nChunks, per);

  knn_combine<<<(B + 255) / 256, 256, 0, stream>>>(partial, out, B);
}
